// ALGNN_20306605376097
// MI455X (gfx1250) — matrix operand order repaired and checked by reading, not compiled, not run on hardware
//
#include <hip/hip_runtime.h>
#include <hip/hip_bf16.h>

typedef __attribute__((ext_vector_type(16))) __bf16 v16bf;
typedef __attribute__((ext_vector_type(8)))  float  v8f;

#define DIMF 128
#define NGRAPH 256

static __device__ __forceinline__ __bf16 f2bf(float f) {
    unsigned u = __builtin_bit_cast(unsigned int, f);
    unsigned r = (u + 0x7FFFu + ((u >> 16) & 1u)) >> 16;   // round-nearest-even
    unsigned short s = (unsigned short)r;
    return __builtin_bit_cast(__bf16, s);
}
static __device__ __forceinline__ float sigmf(float x) { return 1.f / (1.f + __expf(-x)); }
static __device__ __forceinline__ float siluf(float x) { return x * sigmf(x); }

// ---------------------------------------------------------------------------
// LDS-staged WMMA GEMM: C[M,Ncols] = A[M,Kact](f32) @ B[KS*32, Ncols](bf16).
// Block = 256 thr (8 waves) -> 128x128 output tile. A and B tiles are staged
// into LDS in *fragment-ready* per-lane order, so the inner loop is pure
// ds_load_b128 + v_wmma. epi: 1 = +bias, 2 = +bias then SiLU.
// ---------------------------------------------------------------------------
template <int KS>
__global__ __launch_bounds__(256)
void gemm_wmma(const float* __restrict__ A, int M, int Kact,
               const __bf16* __restrict__ B, int Ncols,
               float* __restrict__ C, const float* __restrict__ bias, int epi)
{
    __shared__ alignas(32) __bf16 lA[8 * KS * 32 * 16];   // [wave][ks][lane][16]
    __shared__ alignas(32) __bf16 lB[KS * 8 * 32 * 16];   // [ks][nt][lane][16]

    const int tid     = threadIdx.x;
    const int rowBlk  = blockIdx.x * 128;
    const int colBase = blockIdx.y * 128;
    const int Kpad    = KS * 32;

    // ---- stage A tile: coalesced along K, f32 -> bf16, fragment order ----
    // A-frag layout (16x32 bf16): lane = hi*16 + row%16; half t<8 holds
    // K = ks*32 + hi*8 + t, half t>=8 holds K = ks*32 + hi*8 + 16 + (t-8).
    for (int idx = tid; idx < 128 * Kpad; idx += 256) {
        int row = idx / Kpad, k = idx - row * Kpad;
        int grow = rowBlk + row;
        int gr = grow < M ? grow : M - 1;                 // clamp; junk rows never stored
        float v = (k < Kact) ? A[(long)gr * Kact + k] : 0.f;
        int ks = k >> 5, krel = k & 31;
        int hi = (krel >> 3) & 1;
        int t  = (krel & 7) + ((krel >> 4) << 3);
        int lane = hi * 16 + (row & 15);
        int wv = row >> 4;
        lA[(((wv * KS) + ks) * 32 + lane) * 16 + t] = f2bf(v);
    }
    // ---- stage B tile: coalesced along columns, fragment order ----
    // B-frag layout (32x16 bf16): lane = hi*16 + col%16; half t<8 holds
    // K = ks*32 + hi*8 + t, half t>=8 holds K = ks*32 + hi*8 + 16 + (t-8).
    for (int idx = tid; idx < 128 * Kpad; idx += 256) {
        int k = idx >> 7, c = idx & 127;
        __bf16 v = B[(long)k * Ncols + colBase + c];
        int ks = k >> 5, krel = k & 31;
        int hi = (krel >> 3) & 1, t = (krel & 7) + ((krel >> 4) << 3);
        int nt = c >> 4, l15 = c & 15;
        lB[(((ks * 8) + nt) * 32 + hi * 16 + l15) * 16 + t] = v;
    }
    __syncthreads();

    const int wave = tid >> 5, lane = tid & 31;
    const int hi = lane >> 4, l15 = lane & 15;

    v8f acc[8] = {};
    const v16bf* pA = (const v16bf*)&lA[(wave * KS * 32 + lane) * 16];
    const v16bf* pB = (const v16bf*)&lB[lane * 16];
    #pragma unroll
    for (int ks = 0; ks < KS; ++ks) {
        v16bf a = pA[ks * 32];
        #pragma unroll
        for (int nt = 0; nt < 8; ++nt) {
            v16bf b = pB[(ks * 8 + nt) * 32];
            acc[nt] = __builtin_amdgcn_wmma_f32_16x16x32_bf16(
                false, a, false, b, (short)0, acc[nt], false, false);
        }
    }

    // ---- epilogue: C layout VGPR r -> row rowBase + r + hi*8, col l15 ----
    const int rowBase = rowBlk + wave * 16;
    #pragma unroll
    for (int nt = 0; nt < 8; ++nt) {
        int col = colBase + nt * 16 + l15;
        float bv = (epi >= 1) ? bias[col] : 0.f;
        float* crow = C + (long)(rowBase + hi * 8) * Ncols + col;
        #pragma unroll
        for (int r = 0; r < 8; ++r) {
            int row = rowBase + r + hi * 8;
            if (row < M) {
                float v = acc[nt][r] + bv;
                if (epi == 2) v = siluf(v);
                crow[(long)r * Ncols] = v;
            }
        }
    }
}

// ---------------------------------------------------------------------------
// Weight / bias packing kernels
// ---------------------------------------------------------------------------
__global__ void prep_pad_bf16(const float* __restrict__ W, __bf16* __restrict__ out,
                              int Kact, int Kpad, int Ncols) {
    int idx = blockIdx.x * 256 + threadIdx.x;
    if (idx >= Kpad * Ncols) return;
    int k = idx / Ncols, n = idx % Ncols;
    out[idx] = f2bf(k < Kact ? W[(long)k * Ncols + n] : 0.f);
}
__global__ void prep_wcat(const float* __restrict__ Wsrc, const float* __restrict__ Wdst,
                          const float* __restrict__ We, __bf16* __restrict__ Wcat) {
    int idx = blockIdx.x * 256 + threadIdx.x;           // 128*512
    if (idx >= 128 * 512) return;
    int k = idx >> 9, c = idx & 511;
    float v;
    if      (c < 128) v = Wdst[k * 128 + c];
    else if (c < 256) v = We[k * 128 + (c - 128)];          // We rows   0-127 (x_i block)
    else if (c < 384) v = We[(128 + k) * 128 + (c - 256)];  // We rows 128-255 (x_j block)
    else              v = Wsrc[k * 128 + (c - 384)];
    Wcat[idx] = f2bf(v);
}
__global__ void prep_we3(const float* __restrict__ We, __bf16* __restrict__ We3) {
    int idx = blockIdx.x * 256 + threadIdx.x;           // 128*128
    if (idx >= 128 * 128) return;
    int k = idx >> 7, n = idx & 127;
    We3[idx] = f2bf(We[(256 + k) * 128 + n]);           // We rows 256-383 (e block)
}
__global__ void prep_bias512(const float* __restrict__ bdst, const float* __restrict__ bsrc,
                             float* __restrict__ b512) {
    int c = blockIdx.x * 256 + threadIdx.x;
    if (c >= 512) return;
    b512[c] = (c < 128) ? bdst[c] : (c >= 384 ? bsrc[c - 384] : 0.f);
}
__global__ void zero_f32(float* __restrict__ p, long n) {
    long i = (long)blockIdx.x * 256 + threadIdx.x;
    if (i < n) p[i] = 0.f;
}

// ---------------------------------------------------------------------------
// Gated-GCN edge kernels
// ---------------------------------------------------------------------------
__global__ void edge_sigsum(const float* __restrict__ e, const int* __restrict__ ii,
                            long E128, float* __restrict__ esum) {
    long idx = (long)blockIdx.x * 256 + threadIdx.x;
    if (idx >= E128) return;
    long k = idx >> 7; int f = (int)(idx & 127);
    atomicAdd(&esum[(long)ii[k] * DIMF + f], sigmf(e[idx]));
}

// one wave per edge: gate+msg scatter, LN+SiLU edge residual
__global__ __launch_bounds__(256)
void edge_update(float* __restrict__ e, const int* __restrict__ ii, const int* __restrict__ jj,
                 int E, const float* __restrict__ T, const float* __restrict__ T4,
                 const float* __restrict__ esum, float* __restrict__ agg,
                 const float* __restrict__ g, const float* __restrict__ b) {
    int wave = threadIdx.x >> 5, lane = threadIdx.x & 31;
    int k = blockIdx.x * 8 + wave;
    if (k >= E) return;
    int i = ii[k], j = jj[k];
    const float* Ti = T + (long)i * 512;
    const float* Tj = T + (long)j * 512;
    const float* t4 = T4 + (long)k * DIMF;
    float z[4], eo[4], s = 0.f;
    #pragma unroll
    for (int q = 0; q < 4; ++q) {
        int f = lane + q * 32;
        z[q]  = Ti[128 + f] + Tj[256 + f] + t4[f];   // x_i@We1 + x_j@We2 + e@We3 + be
        eo[q] = e[(long)k * DIMF + f];
        s += z[q];
    }
    for (int o = 16; o; o >>= 1) s += __shfl_xor(s, o, 32);
    float m = s * (1.f / 128.f), v = 0.f;
    #pragma unroll
    for (int q = 0; q < 4; ++q) { float d = z[q] - m; v += d * d; }
    for (int o = 16; o; o >>= 1) v += __shfl_xor(v, o, 32);
    float rstd = rsqrtf(v * (1.f / 128.f) + 1e-5f);
    #pragma unroll
    for (int q = 0; q < 4; ++q) {
        int f = lane + q * 32;
        float ln   = g[f] * (z[q] - m) * rstd + b[f];
        float gate = sigmf(eo[q]) / (esum[(long)i * DIMF + f] + 1e-5f);
        atomicAdd(&agg[(long)i * DIMF + f], gate * Tj[f]);   // gate * (x_j@Wdst+bdst)
        e[(long)k * DIMF + f] = eo[q] + siluf(ln);
    }
}

// one wave per node: x += silu(LN(x@Wsrc + bsrc + agg))
__global__ __launch_bounds__(256)
void node_update(float* __restrict__ x, const float* __restrict__ T,
                 const float* __restrict__ agg, const float* __restrict__ g,
                 const float* __restrict__ b, int N) {
    int wave = threadIdx.x >> 5, lane = threadIdx.x & 31;
    int n = blockIdx.x * 8 + wave;
    if (n >= N) return;
    float z[4], s = 0.f;
    #pragma unroll
    for (int q = 0; q < 4; ++q) {
        int f = lane + q * 32;
        z[q] = T[(long)n * 512 + 384 + f] + agg[(long)n * DIMF + f];
        s += z[q];
    }
    for (int o = 16; o; o >>= 1) s += __shfl_xor(s, o, 32);
    float m = s * (1.f / 128.f), v = 0.f;
    #pragma unroll
    for (int q = 0; q < 4; ++q) { float d = z[q] - m; v += d * d; }
    for (int o = 16; o; o >>= 1) v += __shfl_xor(v, o, 32);
    float rstd = rsqrtf(v * (1.f / 128.f) + 1e-5f);
    #pragma unroll
    for (int q = 0; q < 4; ++q) {
        int f = lane + q * 32;
        float ln = g[f] * (z[q] - m) * rstd + b[f];
        x[(long)n * DIMF + f] += siluf(ln);
    }
}

// ---------------------------------------------------------------------------
// Final stage
// ---------------------------------------------------------------------------
__global__ void bn_reduce(const float* __restrict__ h, int N, float* __restrict__ stats) {
    int f = threadIdx.x;                 // 128 threads
    float s = 0.f, q = 0.f;
    for (int r = blockIdx.x; r < N; r += gridDim.x) {
        float v = h[(long)r * DIMF + f];
        s += v; q += v * v;
    }
    atomicAdd(&stats[f], s);
    atomicAdd(&stats[128 + f], q);
}
__global__ void bn_apply(const float* __restrict__ h, const float* __restrict__ stats,
                         const float* __restrict__ g, const float* __restrict__ b,
                         float* __restrict__ out, int N) {
    long idx = (long)blockIdx.x * 256 + threadIdx.x;
    if (idx >= (long)N * DIMF) return;
    int f = (int)(idx & 127);
    float m   = stats[f] / N;
    float var = stats[128 + f] / N - m * m;
    float v = g[f] * (h[idx] - m) * rsqrtf(var + 1e-5f) + b[f];
    out[idx] = siluf(v);
}
__global__ __launch_bounds__(256)
void pool_kernel(const float* __restrict__ p, const int* __restrict__ batch, int N,
                 float* __restrict__ psum, float* __restrict__ pcnt) {
    int wave = threadIdx.x >> 5, lane = threadIdx.x & 31;
    int n = blockIdx.x * 8 + wave;
    if (n >= N) return;
    int bg = batch[n];
    #pragma unroll
    for (int q = 0; q < 4; ++q) {
        int f = lane + q * 32;
        atomicAdd(&psum[(long)bg * DIMF + f], p[(long)n * DIMF + f]);
    }
    if (lane == 0) atomicAdd(&pcnt[bg], 1.f);
}
__global__ __launch_bounds__(256)
void out_kernel(const float* __restrict__ psum, const float* __restrict__ pcnt,
                const float* __restrict__ oW, const float* __restrict__ oB,
                float* __restrict__ out) {
    int wave = threadIdx.x >> 5, lane = threadIdx.x & 31;
    int gidx = blockIdx.x * 8 + wave;
    if (gidx >= NGRAPH) return;
    float c = fmaxf(pcnt[gidx], 1.f);
    float a = 0.f;
    #pragma unroll
    for (int q = 0; q < 4; ++q) {
        int f = lane + q * 32;
        a += (psum[(long)gidx * DIMF + f] / c) * oW[f];
    }
    for (int o = 16; o; o >>= 1) a += __shfl_xor(a, o, 32);
    if (lane == 0) out[gidx] = a + oB[0];
}

// ---------------------------------------------------------------------------
extern "C" void kernel_launch(void* const* d_in, const int* in_sizes, int n_in,
                              void* d_out, int out_size, void* d_ws, size_t ws_size,
                              hipStream_t stream) {
    (void)in_sizes; (void)n_in; (void)out_size; (void)ws_size;
    const int NN = 20000, NE = 320000, NL = 600000;

    const float* x_in = (const float*)d_in[0];
    const float* ea   = (const float*)d_in[1];
    const float* eal  = (const float*)d_in[2];
    const int*   ei   = (const int*)d_in[3];
    const int*   eil  = (const int*)d_in[4];
    const int*   batch= (const int*)d_in[5];
    const float* pre1W=(const float*)d_in[6];  const float* pre1b=(const float*)d_in[7];
    const float* pre2W=(const float*)d_in[8];  const float* pre2b=(const float*)d_in[9];
    const float* pre3W=(const float*)d_in[10]; const float* pre3b=(const float*)d_in[11];
    const float* lWsrc=(const float*)d_in[12]; const float* lbsrc=(const float*)d_in[13];
    const float* lWdst=(const float*)d_in[14]; const float* lbdst=(const float*)d_in[15];
    const float* lWe  =(const float*)d_in[16]; const float* lbe  =(const float*)d_in[17];
    const float* lnxg =(const float*)d_in[18]; const float* lnxb =(const float*)d_in[19];
    const float* lneg =(const float*)d_in[20]; const float* lneb =(const float*)d_in[21];
    const float* aWsrc=(const float*)d_in[22]; const float* absrc=(const float*)d_in[23];
    const float* aWdst=(const float*)d_in[24]; const float* abdst=(const float*)d_in[25];
    const float* aWe  =(const float*)d_in[26]; const float* abe  =(const float*)d_in[27];
    const float* anxg =(const float*)d_in[28]; const float* anxb =(const float*)d_in[29];
    const float* aneg =(const float*)d_in[30]; const float* aneb =(const float*)d_in[31];
    const float* bn_g =(const float*)d_in[32]; const float* bn_b =(const float*)d_in[33];
    const float* postW=(const float*)d_in[34]; const float* postb=(const float*)d_in[35];
    const float* outW =(const float*)d_in[36]; const float* outb =(const float*)d_in[37];

    char* wsb = (char*)d_ws; size_t off = 0;
    auto allocf = [&](size_t n) { float*  p = (float*)(wsb + off);  off += n * 4;              return p; };
    auto allocb = [&](size_t n) { __bf16* p = (__bf16*)(wsb + off); off += ((n * 2 + 3) & ~3ull); return p; };

    float* h_node = allocf((size_t)NN * DIMF);
    float* h_edge = allocf((size_t)NE * DIMF);
    float* h_line = allocf((size_t)NL * DIMF);
    float* T      = allocf((size_t)NE * 512);     // node-side projections (max M = 320K)
    float* T4     = allocf((size_t)NL * DIMF);    // edge-side e@We3 (max E = 600K)
    float* esum   = allocf((size_t)NE * DIMF);
    float* agg    = allocf((size_t)NE * DIMF);
    float* bias512= allocf(512);
    float* stats  = allocf(256);
    float* psum   = allocf((size_t)NGRAPH * DIMF);
    float* pcnt   = allocf(NGRAPH);
    __bf16* WcatB = allocb(128 * 512);
    __bf16* We3B  = allocb(128 * 128);

    // ---- pre-FC: h = silu(in @ W + b), bf16 WMMA with zero-padded K ----
    prep_pad_bf16<<<(96 * 128 + 255) / 256, 256, 0, stream>>>(pre1W, WcatB, 92, 96, 128);
    gemm_wmma<3><<<dim3((NN + 127) / 128, 1), 256, 0, stream>>>(x_in, NN, 92, WcatB, 128, h_node, pre1b, 2);
    prep_pad_bf16<<<(64 * 128 + 255) / 256, 256, 0, stream>>>(pre2W, WcatB, 50, 64, 128);
    gemm_wmma<2><<<dim3((NE + 127) / 128, 1), 256, 0, stream>>>(ea, NE, 50, WcatB, 128, h_edge, pre2b, 2);
    prep_pad_bf16<<<(64 * 128 + 255) / 256, 256, 0, stream>>>(pre3W, WcatB, 40, 64, 128);
    gemm_wmma<2><<<dim3((NL + 127) / 128, 1), 256, 0, stream>>>(eal, NL, 40, WcatB, 128, h_line, pre3b, 2);

    auto conv = [&](float* x, int Nn, float* e, int Ne, const int* idx,
                    const float* Wsrc, const float* bsrc, const float* Wdst, const float* bdst,
                    const float* We, const float* be, const float* nxg, const float* nxb,
                    const float* neg, const float* neb) {
        const int* ii = idx; const int* jj = idx + Ne;
        long nz = (long)Nn * DIMF;
        zero_f32<<<(unsigned)((nz + 255) / 256), 256, 0, stream>>>(esum, nz);
        zero_f32<<<(unsigned)((nz + 255) / 256), 256, 0, stream>>>(agg, nz);
        prep_wcat<<<(128 * 512 + 255) / 256, 256, 0, stream>>>(Wsrc, Wdst, We, WcatB);
        prep_we3<<<(128 * 128 + 255) / 256, 256, 0, stream>>>(We, We3B);
        prep_bias512<<<2, 256, 0, stream>>>(bdst, bsrc, bias512);
        gemm_wmma<4><<<dim3((Nn + 127) / 128, 4), 256, 0, stream>>>(x, Nn, 128, WcatB, 512, T, bias512, 1);
        gemm_wmma<4><<<dim3((Ne + 127) / 128, 1), 256, 0, stream>>>(e, Ne, 128, We3B, 128, T4, be, 1);
        long e128 = (long)Ne * DIMF;
        edge_sigsum<<<(unsigned)((e128 + 255) / 256), 256, 0, stream>>>(e, ii, e128, esum);
        edge_update<<<(Ne + 7) / 8, 256, 0, stream>>>(e, ii, jj, Ne, T, T4, esum, agg, neg, neb);
        node_update<<<(Nn + 7) / 8, 256, 0, stream>>>(x, T, agg, nxg, nxb, Nn);
    };

    for (int l = 0; l < 4; ++l) {
        size_t o1 = (size_t)l * 128, oW = (size_t)l * 128 * 128, oE = (size_t)l * 384 * 128;
        conv(h_edge, NE, h_line, NL, eil,
             lWsrc + oW, lbsrc + o1, lWdst + oW, lbdst + o1, lWe + oE, lbe + o1,
             lnxg + o1, lnxb + o1, lneg + o1, lneb + o1);
        conv(h_node, NN, h_edge, NE, ei,
             aWsrc + oW, absrc + o1, aWdst + oW, abdst + o1, aWe + oE, abe + o1,
             anxg + o1, anxb + o1, aneg + o1, aneb + o1);
    }

    // ---- BatchNorm over nodes + post-FC + pooling + readout ----
    zero_f32<<<1, 256, 0, stream>>>(stats, 256);
    bn_reduce<<<128, 128, 0, stream>>>(h_node, NN, stats);
    bn_apply<<<(unsigned)(((long)NN * DIMF + 255) / 256), 256, 0, stream>>>(h_node, stats, bn_g, bn_b, esum, NN);
    prep_pad_bf16<<<(128 * 128 + 255) / 256, 256, 0, stream>>>(postW, WcatB, 128, 128, 128);
    gemm_wmma<4><<<dim3((NN + 127) / 128, 1), 256, 0, stream>>>(esum, NN, 128, WcatB, 128, agg, postb, 2);
    zero_f32<<<(NGRAPH * DIMF + NGRAPH + 255) / 256, 256, 0, stream>>>(psum, NGRAPH * DIMF + NGRAPH);
    pool_kernel<<<(NN + 7) / 8, 256, 0, stream>>>(agg, batch, NN, psum, pcnt);
    out_kernel<<<32, 256, 0, stream>>>(psum, pcnt, outW, outb, (float*)d_out);
}
